// StructuralAttentionLayer_61117384622470
// MI455X (gfx1250) — matrix operand order repaired and checked by reading, not compiled, not run on hardware
//
#include <hip/hip_runtime.h>
#include <hip/hip_bf16.h>

#define N_NODES 50000
#define E_EDGES 800000
#define DMODEL  128
#define NHEAD   8
#define DFF     256

typedef __attribute__((ext_vector_type(16))) __bf16 v16bf;
typedef __attribute__((ext_vector_type(8)))  float  v8f;
typedef __attribute__((ext_vector_type(4)))  unsigned int v4u;
typedef __attribute__((ext_vector_type(8)))  int    v8i;
typedef __attribute__((ext_vector_type(4)))  int    v4i;

// ---------------- CDNA5 feature detection -------------------------------
#if defined(__has_builtin)
#if __has_builtin(__builtin_amdgcn_tensor_load_to_lds) && \
    __has_builtin(__builtin_amdgcn_s_wait_tensorcnt)
#define USE_TDM 1
#endif
#endif
#ifndef USE_TDM
#define USE_TDM 0
#endif

__device__ __forceinline__ v8f wmma_bf16(v16bf a, v16bf b, v8f c) {
  return __builtin_amdgcn_wmma_f32_16x16x32_bf16(
      false, a, false, b, (short)0, c, false, false);
}

#if USE_TDM
// 1-D TDM copy: ndw dwords, global -> LDS (D# per CDNA5 ISA).
// This toolchain's builtin is the 6-arg form:
//   (uint32x4 g0, int32x8 g1, int32x4 g2, int32x4 g3, int32x8 gx, i32 cpol)
__device__ __forceinline__ void tdm_load_1d(unsigned lds_off, const void* g,
                                            unsigned ndw) {
  unsigned long long ga = (unsigned long long)g;
  v4u g0;
  g0[0] = 1u;                                   // count=1, user mode
  g0[1] = lds_off;                              // lds_addr (bytes)
  g0[2] = (unsigned)ga;                         // global_addr[31:0]
  g0[3] = (unsigned)((ga >> 32) & 0x1FFFFFFu) | (2u << 30);  // addr hi | type=2
  v8i g1;
  g1[0] = (int)(2u << 16);                      // wg_mask=0, data_size=4B
  g1[1] = (int)(ndw << 16);                     // tensor_dim0[15:0]
  g1[2] = (int)((ndw >> 16) | (1u << 16));      // tensor_dim0 hi | tensor_dim1=1
  g1[3] = (int)(ndw << 16);                     // tensor_dim1 hi=0 | tile_dim0
  g1[4] = 1;                                    // tile_dim1=1, tile_dim2=0
  g1[5] = (int)ndw;                             // tensor_dim0_stride lo
  g1[6] = 0;
  g1[7] = 0;
  v4i g2;
  g2[0] = 1; g2[1] = 0; g2[2] = 0; g2[3] = 0;   // tensor_dim2=1
  v4i g3;
  g3[0] = 0; g3[1] = 0; g3[2] = 0; g3[3] = 0;
  v8i gx;
  gx[0] = 0; gx[1] = 0; gx[2] = 0; gx[3] = 0;
  gx[4] = 0; gx[5] = 0; gx[6] = 0; gx[7] = 0;
  __builtin_amdgcn_tensor_load_to_lds(g0, g1, g2, g3, gx, 0);
}
#endif

// order-preserving f32 <-> u32 for atomicMax-based segment max
__device__ __forceinline__ unsigned ford(float f) {
  int i = __float_as_int(f);
  return (i >= 0) ? ((unsigned)i ^ 0x80000000u) : ~(unsigned)i;
}
__device__ __forceinline__ float funord(unsigned u) {
  int i = (u & 0x80000000u) ? (int)(u ^ 0x80000000u) : (int)(~u);
  return __int_as_float(i);
}

// ---------------------------------------------------------------- init
__global__ __launch_bounds__(256)
void init_kernel(const float* __restrict__ x, float* __restrict__ hbuf,
                 unsigned* __restrict__ segmax, float* __restrict__ segsum) {
  long long t = (long long)blockIdx.x * 256 + threadIdx.x;
  if (t < (long long)N_NODES * DMODEL) hbuf[t] = x[t];
  if (t < (long long)N_NODES * NHEAD) { segmax[t] = 0u; segsum[t] = 0.0f; }
}

// ------------------- weights -> transposed bf16 copies (one-time) -------
__global__ __launch_bounds__(256)
void wconv_kernel(const float* __restrict__ Wq, const float* __restrict__ Wk,
                  const float* __restrict__ Wv, const float* __restrict__ W1,
                  const float* __restrict__ W2, __bf16* __restrict__ WqT,
                  __bf16* __restrict__ WkT, __bf16* __restrict__ WvT,
                  __bf16* __restrict__ W1T, __bf16* __restrict__ W2T) {
  int t = blockIdx.x * 256 + threadIdx.x;
  if (t < 3 * 16384) {                     // Wq/Wk/Wv: [128][128] -> WT[c][r]
    int which = t / 16384, idx = t % 16384;
    int r = idx >> 7, c = idx & 127;
    const float* W = (which == 0) ? Wq : (which == 1) ? Wk : Wv;
    __bf16* WT = (which == 0) ? WqT : (which == 1) ? WkT : WvT;
    WT[c * DMODEL + r] = (__bf16)W[idx];
  } else if (t < 3 * 16384 + 32768) {      // W1: [128][256] -> W1T[c][r]
    int idx = t - 3 * 16384;
    int r = idx >> 8, c = idx & 255;
    W1T[c * DMODEL + r] = (__bf16)W1[idx];
  } else if (t < 3 * 16384 + 65536) {      // W2: [256][128] -> W2T[c][r]
    int idx = t - (3 * 16384 + 32768);
    int r = idx >> 7, c = idx & 127;
    W2T[c * DFF + r] = (__bf16)W2[idx];
  }
}

// ------------------------------------------------- QKV projection (WMMA)
__global__ __launch_bounds__(256)
void qkv_kernel(const float* __restrict__ x, const __bf16* __restrict__ WqT,
                const __bf16* __restrict__ WkT, const __bf16* __restrict__ WvT,
                const float* __restrict__ bq, const float* __restrict__ bk,
                const float* __restrict__ bv, float* __restrict__ Qo,
                float* __restrict__ Ko, float* __restrict__ Vo) {
  __shared__ __bf16 xs[16][DMODEL];                 // 4 KB
#if USE_TDM
  __shared__ __align__(16) float xstage[16 * DMODEL];  // 8 KB staging
#endif
  const int tile = blockIdx.x;
  const int which = blockIdx.y;
  const __bf16* WT   = (which == 0) ? WqT : (which == 1) ? WkT : WvT;
  const float*  bias = (which == 0) ? bq : (which == 1) ? bk : bv;
  float*        out  = (which == 0) ? Qo : (which == 1) ? Ko : Vo;
  const int tid = threadIdx.x;

#if USE_TDM
  if (tid < 32) {  // wave 0: async DMA of the contiguous 16x128 f32 tile
    tdm_load_1d((unsigned)(unsigned long long)&xstage[0],
                x + (size_t)tile * 16 * DMODEL, 16 * DMODEL);
    __builtin_amdgcn_s_wait_tensorcnt(0);
  }
  __syncthreads();
  for (int i = tid; i < 16 * DMODEL; i += 256)
    xs[i >> 7][i & 127] = (__bf16)xstage[i];
#else
  for (int i = tid; i < 16 * DMODEL; i += 256)
    xs[i >> 7][i & 127] =
        (__bf16)x[(size_t)(tile * 16 + (i >> 7)) * DMODEL + (i & 127)];
#endif
  __syncthreads();

  const int lane = tid & 31;
  const int wv   = tid >> 5;   // wave -> 16-col output tile
  const int m    = lane & 15;
  const int half = lane >> 4;
  const int col  = wv * 16 + m;
  v8f c = {};
#pragma unroll
  for (int kt = 0; kt < 4; ++kt) {
    v16bf a;
#pragma unroll
    for (int e = 0; e < 16; ++e) {   // A: contiguous k = 16*half + e (matches B)
      int vg = e >> 1, pos = e & 1;
      int k = kt * 32 + half * 16 + e;
      a[e] = xs[m][k];
    }
    // B fragment: 16 contiguous bf16 from transposed weights (global, L2-hot)
    v16bf b = *(const v16bf*)(WT + (size_t)col * DMODEL + kt * 32 + half * 16);
    c = wmma_bf16(a, b, c);
  }
  const float bb = bias[col];
#pragma unroll
  for (int r = 0; r < 8; ++r) {
    int row = tile * 16 + half * 8 + r;
    out[(size_t)row * DMODEL + col] = c[r] + bb;
  }
}

// ------------------------------------------- edge scores + segment max
__global__ __launch_bounds__(256)
void edge_score_kernel(const int* __restrict__ ei, const float* __restrict__ Q,
                       const float* __restrict__ Kf, float* __restrict__ att,
                       unsigned* __restrict__ segmax) {
  const int tid = threadIdx.x;
  const int lane = tid & 31;
  const long long e = (long long)blockIdx.x * 8 + (tid >> 5);
  if (e >= E_EDGES) return;
  const int src = ei[e];
  const int dst = ei[E_EDGES + e];
  const int off = lane * 4;  // 4 floats/lane; 4 lanes/head
  const float4 q4 = *(const float4*)(Q + (size_t)dst * DMODEL + off);
  const float4 k4 = *(const float4*)(Kf + (size_t)src * DMODEL + off);
  float s = q4.x * k4.x + q4.y * k4.y + q4.z * k4.z + q4.w * k4.w;
  s += __shfl_xor(s, 1, 32);
  s += __shfl_xor(s, 2, 32);
  if ((lane & 3) == 0) {
    int h = lane >> 2;
    float sc = s * 0.25f;  // / sqrt(16)
    att[e * NHEAD + h] = sc;
    atomicMax(segmax + (size_t)dst * NHEAD + h, ford(sc));
  }
}

// -------------------------------------------------- exp + segment sum
__global__ __launch_bounds__(256)
void edge_exp_kernel(const int* __restrict__ ei, float* __restrict__ att,
                     const unsigned* __restrict__ segmax,
                     float* __restrict__ segsum) {
  long long t = (long long)blockIdx.x * 256 + threadIdx.x;
  if (t >= (long long)E_EDGES * NHEAD) return;
  long long e = t >> 3;
  int h = (int)(t & 7);
  int dst = ei[E_EDGES + e];
  float mx = funord(segmax[(size_t)dst * NHEAD + h]);
  float ex = __expf(att[t] - mx);
  att[t] = ex;
  atomicAdd(segsum + (size_t)dst * NHEAD + h, ex);
}

// -------------------------------------- weighted V gather + scatter-add
__global__ __launch_bounds__(256)
void edge_agg_kernel(const int* __restrict__ ei, const float* __restrict__ att,
                     const float* __restrict__ segsum,
                     const float* __restrict__ V, float* __restrict__ hbuf) {
  const int tid = threadIdx.x;
  const int lane = tid & 31;
  const long long e = (long long)blockIdx.x * 8 + (tid >> 5);
  if (e >= E_EDGES) return;
  const int src = ei[e];
  const int dst = ei[E_EDGES + e];
  const int h = lane >> 2;
  float w = att[e * NHEAD + h] / (segsum[(size_t)dst * NHEAD + h] + 1e-16f);
  const int off = lane * 4;
  const float4 v4 = *(const float4*)(V + (size_t)src * DMODEL + off);
  float* p = hbuf + (size_t)dst * DMODEL + off;
  atomicAdd(p + 0, v4.x * w);
  atomicAdd(p + 1, v4.y * w);
  atomicAdd(p + 2, v4.z * w);
  atomicAdd(p + 3, v4.w * w);
}

// -------------------------------------------- LN + FFN (WMMA) + residual
__global__ __launch_bounds__(256)
void ffn_kernel(const float* __restrict__ hbuf, const float* __restrict__ ln_g,
                const float* __restrict__ ln_b, const __bf16* __restrict__ W1T,
                const float* __restrict__ b1p, const __bf16* __restrict__ W2T,
                const float* __restrict__ b2p, float* __restrict__ out) {
  __shared__ __align__(16) float hs[16][DMODEL];  // 8 KB residual + LN source
  __shared__ __bf16 hn[16][DMODEL];               // 4 KB
  __shared__ __bf16 tt[16][DFF];                  // 8 KB post-GELU
  __shared__ float mu_s[16], rs_s[16];
  const int tile = blockIdx.x;
  const int tid = threadIdx.x;

#if USE_TDM
  if (tid < 32) {  // wave 0: DMA the contiguous 16x128 f32 h-tile into LDS
    tdm_load_1d((unsigned)(unsigned long long)&hs[0][0],
                hbuf + (size_t)tile * 16 * DMODEL, 16 * DMODEL);
    __builtin_amdgcn_s_wait_tensorcnt(0);
  }
#else
  for (int i = tid; i < 16 * DMODEL; i += 256)
    hs[i >> 7][i & 127] =
        hbuf[(size_t)(tile * 16 + (i >> 7)) * DMODEL + (i & 127)];
#endif
  __syncthreads();
  if (tid < 16) {
    float s = 0.f;
    for (int j = 0; j < DMODEL; ++j) s += hs[tid][j];
    float mu = s * (1.0f / DMODEL);
    float v = 0.f;
    for (int j = 0; j < DMODEL; ++j) { float d = hs[tid][j] - mu; v += d * d; }
    mu_s[tid] = mu;
    rs_s[tid] = rsqrtf(v * (1.0f / DMODEL) + 1e-5f);
  }
  __syncthreads();
  for (int i = tid; i < 16 * DMODEL; i += 256) {
    int r = i >> 7, c = i & 127;
    hn[r][c] = (__bf16)((hs[r][c] - mu_s[r]) * rs_s[r] * ln_g[c] + ln_b[c]);
  }
  __syncthreads();

  const int lane = tid & 31;
  const int wv   = tid >> 5;
  const int m    = lane & 15;
  const int half = lane >> 4;

  // GEMM1: hn[16x128] @ W1[128x256]; wave handles col tiles wv and wv+8
  v8f c0 = {}, c1 = {};
  const int col0 = wv * 16 + m;
  const int col1 = 128 + col0;
#pragma unroll
  for (int kt = 0; kt < 4; ++kt) {
    v16bf a;
#pragma unroll
    for (int e = 0; e < 16; ++e) {
      int vg = e >> 1, pos = e & 1;
      int k = kt * 32 + half * 16 + e;
      a[e] = hn[m][k];
    }
    v16bf b0 = *(const v16bf*)(W1T + (size_t)col0 * DMODEL + kt * 32 + half * 16);
    v16bf b1 = *(const v16bf*)(W1T + (size_t)col1 * DMODEL + kt * 32 + half * 16);
    c0 = wmma_bf16(a, b0, c0);
    c1 = wmma_bf16(a, b1, c1);
  }
  {  // bias + exact GELU -> tt (bf16)
    float bb0 = b1p[col0], bb1 = b1p[col1];
#pragma unroll
    for (int r = 0; r < 8; ++r) {
      int row = half * 8 + r;
      float v0 = c0[r] + bb0;
      float v1 = c1[r] + bb1;
      tt[row][col0] = (__bf16)(0.5f * v0 * (1.0f + erff(v0 * 0.70710678f)));
      tt[row][col1] = (__bf16)(0.5f * v1 * (1.0f + erff(v1 * 0.70710678f)));
    }
  }
  __syncthreads();

  // GEMM2: tt[16x256] @ W2[256x128]
  v8f c2 = {};
  const int col = wv * 16 + m;
#pragma unroll
  for (int kt = 0; kt < 8; ++kt) {
    v16bf a;
#pragma unroll
    for (int e = 0; e < 16; ++e) {
      int vg = e >> 1, pos = e & 1;
      int k = kt * 32 + half * 16 + e;
      a[e] = tt[m][k];
    }
    v16bf b = *(const v16bf*)(W2T + (size_t)col * DFF + kt * 32 + half * 16);
    c2 = wmma_bf16(a, b, c2);
  }
  const float bb2 = b2p[col];
#pragma unroll
  for (int r = 0; r < 8; ++r) {
    int row = half * 8 + r;
    out[(size_t)(tile * 16 + row) * DMODEL + col] = hs[row][col] + c2[r] + bb2;
  }
}

extern "C" void kernel_launch(void* const* d_in, const int* in_sizes, int n_in,
                              void* d_out, int out_size, void* d_ws,
                              size_t ws_size, hipStream_t stream) {
  const float* x    = (const float*)d_in[0];
  const int*   ei   = (const int*)d_in[1];
  const float* Wq   = (const float*)d_in[2];
  const float* bq   = (const float*)d_in[3];
  const float* Wk   = (const float*)d_in[4];
  const float* bk   = (const float*)d_in[5];
  const float* Wv   = (const float*)d_in[6];
  const float* bv   = (const float*)d_in[7];
  const float* ln_g = (const float*)d_in[8];
  const float* ln_b = (const float*)d_in[9];
  const float* W1   = (const float*)d_in[10];
  const float* b1   = (const float*)d_in[11];
  const float* W2   = (const float*)d_in[12];
  const float* b2   = (const float*)d_in[13];
  float* out = (float*)d_out;

  // workspace layout (all region sizes are multiples of 64B)
  char* ws = (char*)d_ws;
  const size_t ND = (size_t)N_NODES * DMODEL;  // 6.4M
  const size_t EH = (size_t)E_EDGES * NHEAD;   // 6.4M
  const size_t NH = (size_t)N_NODES * NHEAD;   // 0.4M
  float* Q      = (float*)ws;
  float* K      = Q + ND;
  float* V      = K + ND;
  float* att    = V + ND;
  unsigned* smx = (unsigned*)(att + EH);
  float* ssum   = (float*)(smx + NH);
  float* hbuf   = ssum + NH;
  __bf16* WqT   = (__bf16*)(hbuf + ND);
  __bf16* WkT   = WqT + 16384;
  __bf16* WvT   = WkT + 16384;
  __bf16* W1T   = WvT + 16384;
  __bf16* W2T   = W1T + 32768;

  init_kernel<<<(int)((ND + 255) / 256), 256, 0, stream>>>(x, hbuf, smx, ssum);
  wconv_kernel<<<448, 256, 0, stream>>>(Wq, Wk, Wv, W1, W2, WqT, WkT, WvT,
                                        W1T, W2T);
  qkv_kernel<<<dim3(N_NODES / 16, 3), 256, 0, stream>>>(x, WqT, WkT, WvT, bq,
                                                        bk, bv, Q, K, V);
  edge_score_kernel<<<E_EDGES / 8, 256, 0, stream>>>(ei, Q, K, att, smx);
  edge_exp_kernel<<<(E_EDGES * NHEAD) / 256, 256, 0, stream>>>(ei, att, smx,
                                                               ssum);
  edge_agg_kernel<<<E_EDGES / 8, 256, 0, stream>>>(ei, att, ssum, V, hbuf);
  ffn_kernel<<<N_NODES / 16, 256, 0, stream>>>(hbuf, ln_g, ln_b, W1T, b1, W2T,
                                               b2, out);
}
